// TwoBranch_4776003633569
// MI455X (gfx1250) — matrix operand order repaired and checked by reading, not compiled, not run on hardware
//
#include <hip/hip_runtime.h>
#include <hip/hip_bf16.h>
#include <math.h>

#define BB 8
#define FF 64
#define HH 192
#define WW 192
#define NHEADS 8
#define HWSP (HH*WW)          /* 36864  */
#define FH 192
#define FW 97
#define FHW (FH*FW)           /* 18624  */

typedef __bf16 bf16_t;
typedef __attribute__((ext_vector_type(16))) __bf16 v16bf;
typedef __attribute__((ext_vector_type(8)))  __bf16 v8bf;
typedef __attribute__((ext_vector_type(8)))  float  v8f;

union ABfrag { v16bf v; v8bf h8[2]; };

__device__ __forceinline__ ABfrag abzero() {
  ABfrag z;
#pragma unroll
  for (int i = 0; i < 16; ++i) z.v[i] = (bf16_t)0.0f;
  return z;
}

__device__ __forceinline__ v8f vzero8() {
  v8f z;
#pragma unroll
  for (int i = 0; i < 8; ++i) z[i] = 0.0f;
  return z;
}

// ---------------------------------------------------------------------------
// Weight prepack: OIHW f32 -> [co/16][kstep][n=16][k=32] bf16, K = (kh,kw,ci)
// ---------------------------------------------------------------------------
__global__ void pack_w_kernel(const float* __restrict__ w, bf16_t* __restrict__ dst,
                              int Cout, int Cin, int ks) {
  int total = Cout * Cin * ks * ks;
  int i = blockIdx.x * blockDim.x + threadIdx.x;
  if (i >= total) return;
  int kw = i % ks;
  int kh = (i / ks) % ks;
  int ci = (i / (ks * ks)) % Cin;
  int co = i / (ks * ks * Cin);
  int k  = (kh * ks + kw) * Cin + ci;
  int Kt = ks * ks * Cin / 32;
  dst[((size_t)(co >> 4) * Kt + (k >> 5)) * 512 + (size_t)(co & 15) * 32 + (k & 31)]
      = (bf16_t)w[i];
}

// ---------------------------------------------------------------------------
// NCHW f32 -> NHWC bf16 (with channel offset, for concat)
// ---------------------------------------------------------------------------
__global__ void to_nhwc_kernel(const float* __restrict__ src, bf16_t* __restrict__ dst,
                               int C, int Cdst, int coff, int hw, long long total) {
  long long i = (long long)blockIdx.x * blockDim.x + threadIdx.x;
  if (i >= total) return;
  int p = (int)(i % hw);
  int c = (int)((i / hw) % C);
  int b = (int)(i / ((long long)C * hw));
  dst[((size_t)b * hw + p) * Cdst + coff + c] = (bf16_t)src[i];
}

// ---------------------------------------------------------------------------
// Implicit-GEMM conv, fully templated: v_wmma_f32_16x16x32_bf16.
// One wave per 16-wide spatial tile; 4 accumulators = 64 output channels.
// K order = (kh, kw, ci-block); A from NHWC bf16, B from packed bf16 weights.
// A frag (16x32): lane(m=l&15,h=l>>4): elems0-15 = K 16h..16h+15 contiguous.
// B frag (32x16): lane n=l&15: elems = K 16h..16h+15 contiguous.
// D: lane n, vgpr r -> M = r + 8h  => 8 contiguous floats along W per lane.
// act: 0=none 1=relu 2=leaky(0.1) 3=sigmoid ; optional bias / NCHW residual.
// ---------------------------------------------------------------------------
template <int CIN, int COUT, int KS, int HS, int WS>
__global__ __launch_bounds__(32)
void conv_wmma_kernel(const bf16_t* __restrict__ in_nhwc,
                      const bf16_t* __restrict__ wpack,
                      const float* __restrict__ bias,
                      const float* __restrict__ residual,
                      float* __restrict__ out, int act) {
  constexpr int WT  = (WS + 15) >> 4;
  constexpr int PAD = KS >> 1;
  constexpr int KT  = KS * KS * (CIN >> 5);
  constexpr bool FULLW = (WS % 16) == 0;

  int tile = blockIdx.x;
  int wt = tile % WT;
  int h  = (tile / WT) % HS;
  int b  = tile / (WT * HS);
  int lane = threadIdx.x & 31;
  int m  = lane & 15;
  int hf = lane >> 4;
  int wpos = wt * 16 + m;

#pragma unroll
  for (int cb = 0; cb < COUT; cb += 64) {
    v8f acc[4];
#pragma unroll
    for (int i = 0; i < 4; ++i) acc[i] = vzero8();

#pragma unroll
    for (int kh = 0; kh < KS; ++kh) {
      int hp = h + kh - PAD;
      bool hin = (hp >= 0) && (hp < HS);
#pragma unroll
      for (int kw2 = 0; kw2 < KS; ++kw2) {
        int wp = wpos + kw2 - PAD;
        bool inb = hin && (wp >= 0) && (wp < WS);
        const bf16_t* src = in_nhwc + (((size_t)b * HS + hp) * WS + wp) * CIN;
        const bf16_t* wbase = wpack
            + ((size_t)(cb >> 4) * KT + (kh * KS + kw2) * (CIN >> 5)) * 512
            + (size_t)m * 32 + 16 * hf;
#pragma unroll
        for (int cib = 0; cib < CIN; cib += 32) {
          ABfrag afr;
          if (inb) {
            afr.h8[0] = *(const v8bf*)(src + cib + 16 * hf);       // K = 16h..16h+7
            afr.h8[1] = *(const v8bf*)(src + cib + 16 * hf + 8);   // K = 16h+8..16h+15
          } else {
            afr = abzero();
          }
#pragma unroll
          for (int nt = 0; nt < 4; ++nt) {
            const bf16_t* bp = wbase + ((size_t)nt * KT + (cib >> 5)) * 512;
            ABfrag bfr;
            bfr.h8[0] = *(const v8bf*)(bp);
            bfr.h8[1] = *(const v8bf*)(bp + 8);
            acc[nt] = __builtin_amdgcn_wmma_f32_16x16x32_bf16(
                false, afr.v, false, bfr.v, (short)0, acc[nt], false, false);
          }
        }
      }
    }

#pragma unroll
    for (int nt = 0; nt < 4; ++nt) {
      int c = cb + nt * 16 + m;
      float bv = bias ? bias[c] : 0.0f;
      size_t base = (((size_t)b * COUT + c) * HS + h) * WS + (size_t)wt * 16 + 8 * hf;
      float v[8];
#pragma unroll
      for (int r = 0; r < 8; ++r) {
        float t = acc[nt][r] + bv;
        if (act == 1)      t = fmaxf(t, 0.0f);
        else if (act == 2) t = (t > 0.0f) ? t : 0.1f * t;
        else if (act == 3) t = 1.0f / (1.0f + __expf(-t));
        v[r] = t;
      }
      if (FULLW) {
        // 8 contiguous, 32B-aligned floats -> two b128 stores (+ b128 residual)
        if (residual) {
          float4 r0 = *(const float4*)(residual + base);
          float4 r1 = *(const float4*)(residual + base + 4);
          v[0] += r0.x; v[1] += r0.y; v[2] += r0.z; v[3] += r0.w;
          v[4] += r1.x; v[5] += r1.y; v[6] += r1.z; v[7] += r1.w;
        }
        *(float4*)(out + base)     = make_float4(v[0], v[1], v[2], v[3]);
        *(float4*)(out + base + 4) = make_float4(v[4], v[5], v[6], v[7]);
      } else {
#pragma unroll
        for (int r = 0; r < 8; ++r) {
          int wst = wt * 16 + 8 * hf + r;
          if (wst < WS) {
            float t = v[r];
            if (residual) t += residual[base + r];
            out[base + r] = t;
          }
        }
      }
    }
  }
}

// ---------------------------------------------------------------------------
// Head: 3x3 conv Cin=3 -> 64, ReLU, no bias (direct; tiny K)
// ---------------------------------------------------------------------------
__global__ __launch_bounds__(256)
void head_conv_kernel(const float* __restrict__ x, const float* __restrict__ w,
                      float* __restrict__ out) {
  __shared__ float ws[64 * 3 * 9];
  for (int i = threadIdx.x; i < 64 * 3 * 9; i += 256) ws[i] = w[i];
  __syncthreads();
  long long i = (long long)blockIdx.x * 256 + threadIdx.x;
  if (i >= (long long)BB * 64 * HWSP) return;
  int wq = (int)(i % WW);
  int hq = (int)((i / WW) % HH);
  int co = (int)((i / HWSP) % 64);
  int b  = (int)(i / ((long long)64 * HWSP));
  float s = 0.0f;
  for (int ci = 0; ci < 3; ++ci)
    for (int kh = 0; kh < 3; ++kh) {
      int hp = hq + kh - 1;
      if (hp < 0 || hp >= HH) continue;
      for (int kw = 0; kw < 3; ++kw) {
        int wp = wq + kw - 1;
        if (wp < 0 || wp >= WW) continue;
        s += x[(((size_t)b * 3 + ci) * HH + hp) * WW + wp]
           * ws[((co * 3 + ci) * 3 + kh) * 3 + kw];
      }
    }
  out[i] = fmaxf(s, 0.0f);
}

// ---------------------------------------------------------------------------
// Depthwise 3x3, pad=1 (q/kv dwconv)
// ---------------------------------------------------------------------------
__global__ __launch_bounds__(256)
void dw3x3_kernel(const float* __restrict__ in, const float* __restrict__ w,
                  float* __restrict__ out, int C) {
  long long n = (long long)BB * C * HWSP;
  long long i = (long long)blockIdx.x * 256 + threadIdx.x;
  if (i >= n) return;
  int wq = (int)(i % WW);
  int hq = (int)((i / WW) % HH);
  long long plane = i / HWSP;
  int c = (int)(plane % C);
  size_t base = (size_t)plane * HWSP;
  float s = 0.0f;
#pragma unroll
  for (int kh = 0; kh < 3; ++kh) {
    int hp = hq + kh - 1;
    if (hp < 0 || hp >= HH) continue;
#pragma unroll
    for (int kw = 0; kw < 3; ++kw) {
      int wp = wq + kw - 1;
      if (wp < 0 || wp >= WW) continue;
      s += in[base + (size_t)hp * WW + wp] * w[c * 9 + kh * 3 + kw];
    }
  }
  out[i] = s;
}

// ---------------------------------------------------------------------------
// rfft along W (real 192 -> 97 complex), +1e-8 folded in. LDS twiddles.
// ---------------------------------------------------------------------------
__global__ __launch_bounds__(128)
void fftw_kernel(const float* __restrict__ src, float* __restrict__ Z) {
  __shared__ float row[192], twc[192], tws[192];
  int r = blockIdx.x;                     // b*64*192 rows
  const float* rp = src + (size_t)r * 192;
  for (int i = threadIdx.x; i < 192; i += 128) {
    row[i] = rp[i] + 1e-8f;
    float s, c;
    __sincosf(-2.0f * 3.14159265358979323846f * i / 192.0f, &s, &c);
    twc[i] = c; tws[i] = s;
  }
  __syncthreads();
  for (int k = threadIdx.x; k < 97; k += 128) {
    float re = 0.0f, im = 0.0f;
    for (int w = 0; w < 192; ++w) {
      int t = (w * k) % 192;
      float xv = row[w];
      re += xv * twc[t];
      im += xv * tws[t];
    }
    size_t o = ((size_t)r * 97 + k) * 2;
    Z[o] = re; Z[o + 1] = im;
  }
}

// ---------------------------------------------------------------------------
// Forward complex DFT along H of the 192x97 grid -> amp / phase
// ---------------------------------------------------------------------------
__global__ __launch_bounds__(192)
void ffth_kernel(const float* __restrict__ Z, float* __restrict__ amp,
                 float* __restrict__ pha) {
  __shared__ float cr[192], cimg[192], twc[192], tws[192];
  int col = blockIdx.x % FW;
  int bc  = blockIdx.x / FW;
  int t   = threadIdx.x;
  {
    float s, c;
    __sincosf(-2.0f * 3.14159265358979323846f * t / 192.0f, &s, &c);
    twc[t] = c; tws[t] = s;
    size_t idx = (((size_t)bc * 192 + t) * FW + col) * 2;
    cr[t] = Z[idx]; cimg[t] = Z[idx + 1];
  }
  __syncthreads();
  float re = 0.0f, im = 0.0f;
  for (int hh = 0; hh < 192; ++hh) {
    int tt = (hh * t) % 192;
    float c = twc[tt], s = tws[tt];
    re += cr[hh] * c - cimg[hh] * s;
    im += cr[hh] * s + cimg[hh] * c;
  }
  size_t o = ((size_t)bc * 192 + t) * FW + col;
  amp[o] = sqrtf(re * re + im * im);
  pha[o] = atan2f(im, re);
}

// ---------------------------------------------------------------------------
// real/imag = a*cos(p)+2e-8 / a*sin(p)+1e-8  (the two +1e-8 of the reference)
// ---------------------------------------------------------------------------
__global__ __launch_bounds__(256)
void polar_kernel(const float* __restrict__ a, const float* __restrict__ p,
                  float* __restrict__ Z, long long n) {
  long long i = (long long)blockIdx.x * 256 + threadIdx.x;
  if (i >= n) return;
  float s, c;
  __sincosf(p[i], &s, &c);
  float av = a[i];
  Z[2 * i]     = av * c + 2e-8f;
  Z[2 * i + 1] = av * s + 1e-8f;
}

// ---------------------------------------------------------------------------
// Inverse complex DFT along H (in-place per column; block owns its column)
// ---------------------------------------------------------------------------
__global__ __launch_bounds__(192)
void iffth_kernel(float* __restrict__ Z) {
  __shared__ float cr[192], cimg[192], twc[192], tws[192];
  int col = blockIdx.x % FW;
  int bc  = blockIdx.x / FW;
  int t   = threadIdx.x;
  size_t idx = (((size_t)bc * 192 + t) * FW + col) * 2;
  {
    float s, c;
    __sincosf(2.0f * 3.14159265358979323846f * t / 192.0f, &s, &c);
    twc[t] = c; tws[t] = s;
    cr[t] = Z[idx]; cimg[t] = Z[idx + 1];
  }
  __syncthreads();
  float re = 0.0f, im = 0.0f;
  for (int k = 0; k < 192; ++k) {
    int tt = (k * t) % 192;
    float c = twc[tt], s = tws[tt];
    re += cr[k] * c - cimg[k] * s;
    im += cr[k] * s + cimg[k] * c;
  }
  Z[idx] = re; Z[idx + 1] = im;
}

// ---------------------------------------------------------------------------
// irfft along W (97 complex -> 192 real) with 1/(H*W) and |.|
// ---------------------------------------------------------------------------
__global__ __launch_bounds__(192)
void irfftw_kernel(const float* __restrict__ Z, float* __restrict__ rec) {
  __shared__ float zr[97], zi[97], twc[192], tws[192];
  int r = blockIdx.x;
  int t = threadIdx.x;
  {
    float s, c;
    __sincosf(2.0f * 3.14159265358979323846f * t / 192.0f, &s, &c);
    twc[t] = c; tws[t] = s;
  }
  if (t < 97) {
    size_t idx = ((size_t)r * 97 + t) * 2;
    zr[t] = Z[idx]; zi[t] = Z[idx + 1];
  }
  __syncthreads();
  float acc = zr[0] + ((t & 1) ? -zr[96] : zr[96]);
  for (int k = 1; k < 96; ++k) {
    int tt = (t * k) % 192;
    acc += 2.0f * (zr[k] * twc[tt] - zi[k] * tws[tt]);
  }
  rec[(size_t)r * 192 + t] = fabsf(acc * (1.0f / (192.0f * 192.0f)));
}

// ---------------------------------------------------------------------------
// nan_to_num (in place)
// ---------------------------------------------------------------------------
__global__ __launch_bounds__(256)
void nan2num_kernel(float* __restrict__ x, long long n) {
  long long i = (long long)blockIdx.x * 256 + threadIdx.x;
  if (i >= n) return;
  float v = x[i];
  if (!__builtin_isfinite(v)) x[i] = 1e-5f;
}

// ---------------------------------------------------------------------------
// per-(b,c) L2 norm over HW -> scale = 1/max(||.||, 1e-12)
// ---------------------------------------------------------------------------
__global__ __launch_bounds__(256)
void rownorm_kernel(const float* __restrict__ x, float* __restrict__ scale,
                    int cstride) {
  int row = blockIdx.x;           // b*64 + c
  int b = row >> 6, c = row & 63;
  const float* p = x + ((size_t)b * cstride + c) * HWSP;
  float s = 0.0f;
  for (int i = threadIdx.x; i < HWSP; i += 256) {
    __builtin_prefetch(p + i + 4096, 0, 1);
    float v = p[i];
    s += v * v;
  }
  __shared__ float sm[256];
  sm[threadIdx.x] = s;
  __syncthreads();
  for (int o = 128; o > 0; o >>= 1) {
    if (threadIdx.x < o) sm[threadIdx.x] += sm[threadIdx.x + o];
    __syncthreads();
  }
  if (threadIdx.x == 0) scale[row] = 1.0f / fmaxf(sqrtf(sm[0]), 1e-12f);
}

// ---------------------------------------------------------------------------
// S[b,h,c,d] = <q_c, k_d> * qs_c * ks_d   (8x8 per head, 4 partials/pair)
// ---------------------------------------------------------------------------
__global__ __launch_bounds__(256)
void qk_dot_kernel(const float* __restrict__ q, const float* __restrict__ k,
                   const float* __restrict__ qs, const float* __restrict__ ksc,
                   float* __restrict__ S, int kcst) {
  int bh = blockIdx.x;            // b*8 + h
  int b = bh >> 3, h = bh & 7;
  int t = threadIdx.x;
  int pr = t >> 2, part = t & 3;
  int c = pr >> 3, d = pr & 7;
  const float* qp = q + ((size_t)b * 64 + h * 8 + c) * HWSP;
  const float* kp = k + ((size_t)b * kcst + h * 8 + d) * HWSP;
  float s = 0.0f;
  for (int n = part; n < HWSP; n += 4) {
    __builtin_prefetch(qp + n + 2048, 0, 1);
    __builtin_prefetch(kp + n + 2048, 0, 1);
    s += qp[n] * kp[n];
  }
  __shared__ float sm[256];
  sm[t] = s;
  __syncthreads();
  if (part == 0) {
    float tot = sm[t] + sm[t + 1] + sm[t + 2] + sm[t + 3];
    S[(size_t)bh * 64 + pr] = tot * qs[b * 64 + h * 8 + c] * ksc[b * 64 + h * 8 + d];
  }
}

// ---------------------------------------------------------------------------
// softmax over d (8) with temperature[h]
// ---------------------------------------------------------------------------
__global__ __launch_bounds__(64)
void softmax8_kernel(const float* __restrict__ S, const float* __restrict__ temp,
                     float* __restrict__ A) {
  int bh = blockIdx.x;
  int h = bh & 7;
  int t = threadIdx.x;
  int c = t >> 3;
  __shared__ float sm[64];
  sm[t] = S[(size_t)bh * 64 + t] * temp[h];
  __syncthreads();
  float mx = -1e30f;
#pragma unroll
  for (int j = 0; j < 8; ++j) mx = fmaxf(mx, sm[c * 8 + j]);
  float sum = 0.0f;
#pragma unroll
  for (int j = 0; j < 8; ++j) sum += __expf(sm[c * 8 + j] - mx);
  A[(size_t)bh * 64 + t] = __expf(sm[t] - mx) / sum;
}

// ---------------------------------------------------------------------------
// o[b,c,n] = sum_d A[b,h,c',d] * v[b,voff+h*8+d,n]
// ---------------------------------------------------------------------------
__global__ __launch_bounds__(256)
void attn_apply_kernel(const float* __restrict__ A, const float* __restrict__ v,
                       float* __restrict__ o, int vcst, int voff) {
  long long n = (long long)BB * 64 * HWSP;
  long long i = (long long)blockIdx.x * 256 + threadIdx.x;
  if (i >= n) return;
  int pix = (int)(i % HWSP);
  int c   = (int)((i / HWSP) % 64);
  int b   = (int)(i / ((long long)64 * HWSP));
  int h   = c >> 3;
  const float* ar = A + ((size_t)b * 8 + h) * 64 + (size_t)(c & 7) * 8;
  const float* vp = v + ((size_t)b * vcst + voff + h * 8) * HWSP + pix;
  float s = 0.0f;
#pragma unroll
  for (int d = 0; d < 8; ++d) s += ar[d] * vp[(size_t)d * HWSP];
  o[i] = s;
}

// ---------------------------------------------------------------------------
// res = fre3*gate[:,:64] + spa3*gate[:,64:]; nan_to_num
// ---------------------------------------------------------------------------
__global__ __launch_bounds__(256)
void final_kernel(const float* __restrict__ fre3, const float* __restrict__ spa3,
                  const float* __restrict__ gate, float* __restrict__ out) {
  long long n = (long long)BB * 64 * HWSP;
  long long i = (long long)blockIdx.x * 256 + threadIdx.x;
  if (i >= n) return;
  int pix = (int)(i % HWSP);
  int c   = (int)((i / HWSP) % 64);
  int b   = (int)(i / ((long long)64 * HWSP));
  float fa = gate[((size_t)b * 128 + c) * HWSP + pix];
  float sa = gate[((size_t)b * 128 + 64 + c) * HWSP + pix];
  float v = fre3[i] * fa + spa3[i] * sa;
  if (!__builtin_isfinite(v)) v = 1e-5f;
  out[i] = v;
}

// ---------------------------------------------------------------------------
// host-side templated conv launcher
// ---------------------------------------------------------------------------
template <int CIN, int COUT, int KS, int HS, int WS>
static void launch_conv(const bf16_t* in, const bf16_t* wp, const float* bias,
                        const float* res, float* out, int act, hipStream_t stream) {
  constexpr int WT = (WS + 15) / 16;
  conv_wmma_kernel<CIN, COUT, KS, HS, WS><<<BB * HS * WT, 32, 0, stream>>>(
      in, wp, bias, res, out, act);
}

// ===========================================================================
extern "C" void kernel_launch(void* const* d_in, const int* in_sizes, int n_in,
                              void* d_out, int out_size, void* d_ws, size_t ws_size,
                              hipStream_t stream) {
  (void)in_sizes; (void)n_in; (void)out_size;
  const float* x       = (const float*)d_in[0];
  const float* head_w  = (const float*)d_in[1];
  const float* fpre_w  = (const float*)d_in[2];
  const float* fpre_b  = (const float*)d_in[3];
  const float* amp1_w  = (const float*)d_in[4];
  const float* amp1_b  = (const float*)d_in[5];
  const float* amp2_w  = (const float*)d_in[6];
  const float* amp2_b  = (const float*)d_in[7];
  const float* pha1_w  = (const float*)d_in[8];
  const float* pha1_b  = (const float*)d_in[9];
  const float* pha2_w  = (const float*)d_in[10];
  const float* pha2_b  = (const float*)d_in[11];
  const float* post_w  = (const float*)d_in[12];
  const float* post_b  = (const float*)d_in[13];
  const float* fre_w   = (const float*)d_in[14];
  const float* fre_b   = (const float*)d_in[15];
  const float* spa_w   = (const float*)d_in[16];
  const float* spa_b   = (const float*)d_in[17];
  const float* temp    = (const float*)d_in[18];
  const float* kv_w    = (const float*)d_in[19];
  const float* kvdw_w  = (const float*)d_in[20];
  const float* q_w     = (const float*)d_in[21];
  const float* qdw_w   = (const float*)d_in[22];
  const float* proj_w  = (const float*)d_in[23];
  const float* fuse1_w = (const float*)d_in[24];
  const float* fuse1_b = (const float*)d_in[25];
  const float* fuse2_w = (const float*)d_in[26];
  const float* fuse2_b = (const float*)d_in[27];

  // ---- workspace arena --------------------------------------------------
  char* base = (char*)d_ws;
  size_t off = 0;
  auto alloc = [&](size_t bytes) -> char* {
    char* p = base + off;
    off += (bytes + 255) & ~(size_t)255;
    return p;
  };
  const size_t T64f = (size_t)BB * FF * HWSP * sizeof(float);     // 75.5 MB
  float*  f     = (float*)alloc(T64f);
  float*  tmpA  = (float*)alloc(T64f);
  float*  tmpB  = (float*)alloc(T64f);
  float*  fre   = (float*)alloc(T64f);
  float*  fre2  = (float*)alloc(T64f);
  float*  spa2  = (float*)alloc(T64f);
  float*  fre3  = (float*)alloc(T64f);
  float*  spa3  = (float*)alloc(T64f);
  float*  t128a = (float*)alloc(2 * T64f);
  float*  t128b = (float*)alloc(2 * T64f);
  bf16_t* nhwcA   = (bf16_t*)alloc(T64f / 2);
  bf16_t* nhwc128 = (bf16_t*)alloc(T64f);
  float*  scal  = (float*)alloc(65536);
  bf16_t* wpk   = (bf16_t*)alloc(1 << 20);
  if (off > ws_size) return;   // workspace too small: do nothing (no OOB)

  // aliases (lifetime-disjoint)
  float* fpre = t128b;          // (B,64,H,W) f32
  float* Zc   = t128a;          // complex freq grid, 76.3MB
  float* amp  = tmpA;           // (B,64,192,97)
  float* pha  = tmpB;
  float* fa   = fre2;           // amp-branch output
  float* fp_  = spa2;           // pha-branch output
  float* ft   = fre3;           // freq conv temp / later rec
  float* rec  = fre3;

  float* qs  = scal;            // 512
  float* ksc = scal + 512;      // 512
  float* S   = scal + 1024;     // 4096
  float* Aat = scal + 1024 + 4096;

  size_t wo = 0;
  auto wseg = [&](size_t els) -> bf16_t* { bf16_t* p = wpk + wo; wo += els; return p; };
  bf16_t* pk_fpre = wseg(4096);
  bf16_t* pk_amp1 = wseg(36864);
  bf16_t* pk_amp2 = wseg(36864);
  bf16_t* pk_pha1 = wseg(36864);
  bf16_t* pk_pha2 = wseg(36864);
  bf16_t* pk_post = wseg(4096);
  bf16_t* pk_frew = wseg(36864);
  bf16_t* pk_spaw = wseg(36864);
  bf16_t* pk_kv   = wseg(8192);
  bf16_t* pk_q    = wseg(4096);
  bf16_t* pk_proj = wseg(4096);
  bf16_t* pk_f1   = wseg(73728);
  bf16_t* pk_f2   = wseg(73728);

  auto packw = [&](const float* w, bf16_t* dst, int Cout, int Cin, int ks) {
    int total = Cout * Cin * ks * ks;
    pack_w_kernel<<<(total + 255) / 256, 256, 0, stream>>>(w, dst, Cout, Cin, ks);
  };
  auto tonhwc = [&](const float* src, bf16_t* dst, int C, int Cdst, int coff, int hw) {
    long long total = (long long)BB * C * hw;
    to_nhwc_kernel<<<(unsigned)((total + 255) / 256), 256, 0, stream>>>(
        src, dst, C, Cdst, coff, hw, total);
  };

  // ---- weight prepack ---------------------------------------------------
  packw(fpre_w,  pk_fpre, 64, 64, 1);
  packw(amp1_w,  pk_amp1, 64, 64, 3);
  packw(amp2_w,  pk_amp2, 64, 64, 3);
  packw(pha1_w,  pk_pha1, 64, 64, 3);
  packw(pha2_w,  pk_pha2, 64, 64, 3);
  packw(post_w,  pk_post, 64, 64, 1);
  packw(fre_w,   pk_frew, 64, 64, 3);
  packw(spa_w,   pk_spaw, 64, 64, 3);
  packw(kv_w,    pk_kv,  128, 64, 1);
  packw(q_w,     pk_q,    64, 64, 1);
  packw(proj_w,  pk_proj, 64, 64, 1);
  packw(fuse1_w, pk_f1,   64, 128, 3);
  packw(fuse2_w, pk_f2,  128, 64, 3);

  const long long T64 = (long long)BB * 64 * HWSP;
  const long long FQ  = (long long)BB * 64 * FHW;

  // ---- head -------------------------------------------------------------
  head_conv_kernel<<<(unsigned)((T64 + 255) / 256), 256, 0, stream>>>(x, head_w, f);

  // ---- FreBlock9 --------------------------------------------------------
  tonhwc(f, nhwcA, 64, 64, 0, HWSP);
  launch_conv<64, 64, 1, HH, WW>(nhwcA, pk_fpre, fpre_b, nullptr, fpre, 0, stream);
  fftw_kernel<<<BB * 64 * HH, 128, 0, stream>>>(fpre, Zc);
  ffth_kernel<<<BB * 64 * FW, 192, 0, stream>>>(Zc, amp, pha);
  // amplitude branch
  tonhwc(amp, nhwcA, 64, 64, 0, FHW);
  launch_conv<64, 64, 3, FH, FW>(nhwcA, pk_amp1, amp1_b, nullptr, ft, 2, stream);
  tonhwc(ft, nhwcA, 64, 64, 0, FHW);
  launch_conv<64, 64, 3, FH, FW>(nhwcA, pk_amp2, amp2_b, amp, fa, 0, stream);
  // phase branch
  tonhwc(pha, nhwcA, 64, 64, 0, FHW);
  launch_conv<64, 64, 3, FH, FW>(nhwcA, pk_pha1, pha1_b, nullptr, ft, 2, stream);
  tonhwc(ft, nhwcA, 64, 64, 0, FHW);
  launch_conv<64, 64, 3, FH, FW>(nhwcA, pk_pha2, pha2_b, pha, fp_, 0, stream);
  // recombine + inverse FFT
  polar_kernel<<<(unsigned)((FQ + 255) / 256), 256, 0, stream>>>(fa, fp_, Zc, FQ);
  iffth_kernel<<<BB * 64 * FW, 192, 0, stream>>>(Zc);
  irfftw_kernel<<<BB * 64 * HH, 192, 0, stream>>>(Zc, rec);
  tonhwc(rec, nhwcA, 64, 64, 0, HWSP);
  launch_conv<64, 64, 1, HH, WW>(nhwcA, pk_post, post_b, f, fre, 0, stream);
  nan2num_kernel<<<(unsigned)((T64 + 255) / 256), 256, 0, stream>>>(fre, T64);

  // ---- FuseBlock7 -------------------------------------------------------
  tonhwc(fre, nhwcA, 64, 64, 0, HWSP);
  launch_conv<64, 64, 3, HH, WW>(nhwcA, pk_frew, fre_b, nullptr, fre2, 0, stream);
  tonhwc(f, nhwcA, 64, 64, 0, HWSP);
  launch_conv<64, 64, 3, HH, WW>(nhwcA, pk_spaw, spa_b, nullptr, spa2, 0, stream);

  auto attn = [&](const float* xq, const float* ykv, const float* resid, float* outb) {
    // kv = dw3x3(conv1x1(ykv))
    tonhwc(ykv, nhwcA, 64, 64, 0, HWSP);
    launch_conv<64, 128, 1, HH, WW>(nhwcA, pk_kv, nullptr, nullptr, t128a, 0, stream);
    dw3x3_kernel<<<(unsigned)(((long long)BB * 128 * HWSP + 255) / 256), 256, 0,
                   stream>>>(t128a, kvdw_w, t128b, 128);
    // q = dw3x3(conv1x1(xq))
    tonhwc(xq, nhwcA, 64, 64, 0, HWSP);
    launch_conv<64, 64, 1, HH, WW>(nhwcA, pk_q, nullptr, nullptr, tmpA, 0, stream);
    dw3x3_kernel<<<(unsigned)((T64 + 255) / 256), 256, 0, stream>>>(
        tmpA, qdw_w, tmpB, 64);
    // normalized channel gram + softmax
    rownorm_kernel<<<BB * 64, 256, 0, stream>>>(tmpB, qs, 64);
    rownorm_kernel<<<BB * 64, 256, 0, stream>>>(t128b, ksc, 128);
    qk_dot_kernel<<<BB * NHEADS, 256, 0, stream>>>(tmpB, t128b, qs, ksc, S, 128);
    softmax8_kernel<<<BB * NHEADS, 64, 0, stream>>>(S, temp, Aat);
    attn_apply_kernel<<<(unsigned)((T64 + 255) / 256), 256, 0, stream>>>(
        Aat, t128b, tmpA, 128, 64);
    // project (+ residual)
    tonhwc(tmpA, nhwcA, 64, 64, 0, HWSP);
    launch_conv<64, 64, 1, HH, WW>(nhwcA, pk_proj, nullptr, resid, outb, 0, stream);
  };
  attn(fre2, spa2, fre2, fre3);   // fre3 = attn(fre2, spa2) + fre2
  attn(spa2, fre3, spa2, spa3);   // spa3 = attn(spa2, fre3) + spa2

  // ---- fuse gate --------------------------------------------------------
  tonhwc(fre3, nhwc128, 64, 128, 0,  HWSP);
  tonhwc(spa3, nhwc128, 64, 128, 64, HWSP);
  launch_conv<128, 64, 3, HH, WW>(nhwc128, pk_f1, fuse1_b, nullptr, tmpA, 0, stream);
  tonhwc(tmpA, nhwcA, 64, 64, 0, HWSP);
  launch_conv<64, 128, 3, HH, WW>(nhwcA, pk_f2, fuse2_b, nullptr, t128a, 3, stream);
  final_kernel<<<(unsigned)((T64 + 255) / 256), 256, 0, stream>>>(
      fre3, spa3, t128a, (float*)d_out);
}
